// trajectory2seq_39135742001438
// MI455X (gfx1250) — matrix operand order repaired and checked by reading, compiled, not run on hardware
//
#include <hip/hip_runtime.h>
#include <math.h>

// Problem dims (match reference)
#define H   512
#define DD  32000
#define SS  512
#define TT  32
#define BB  64

typedef __bf16 bf16;
typedef __attribute__((ext_vector_type(16))) __bf16 v16bf;
typedef __attribute__((ext_vector_type(8)))  float  v8f;

union Frag16 { v16bf v; unsigned int u[8]; };

// A-fragment: 16x32 bf16 tile; `row_base` already points at this lane's row.
// Layout: lanes 0-15 = rows, vgpr j holds a K pair; lanes 0-15 hold K=0..15,
// lanes 16-31 hold K=16..31 (matches B fragment).
__device__ inline void load_a_frag_row(Frag16& f, const bf16* row_base, int half) {
#pragma unroll
    for (int j = 0; j < 8; ++j) {
        const int kk = half * 16 + (j << 1);
        f.u[j] = *reinterpret_cast<const unsigned int*>(row_base + kk);
    }
}

// B-fragment: 32x16 bf16 (KxN) from W[N,K] row-major (B[k][n] = W[n][k]).
// lanes 0-15 hold K=0..15 (pairs per vgpr), lanes 16-31 hold K=16..31.
// `col_base` already points at W[n0+col] row + half*16.
__device__ inline void load_b_frag_row(Frag16& f, const bf16* col_base) {
#pragma unroll
    for (int j = 0; j < 8; ++j)
        f.u[j] = *reinterpret_cast<const unsigned int*>(col_base + j * 2);
}

__device__ inline v8f wmma_bf16(const Frag16& a, const Frag16& b, v8f c) {
    return __builtin_amdgcn_wmma_f32_16x16x32_bf16(false, a.v, false, b.v,
                                                   (short)0, c, false, false);
}

__device__ inline float sigm(float x) { return 1.f / (1.f + __expf(-x)); }

// ---------------------------------------------------------------------------
// Fused GRU gate kernel: gi = A @ Wih^T + bih ; gh = h @ Whh^T + bhh ;
// r,z,n gates + state update, all in-register on WMMA fragments.
// Block: 128 threads (4 waves, one 16-row M tile each). Grid: H/16 column tiles.
// A source: (a) bf16 [B,Kin] matrix, optionally row-gathered via tok (decoder
// embedding lookup), or (b) f32 [B,2] elementwise path (encoder layer 0).
// Inner K-loops are 2-stage software pipelined: next A/B fragments are issued
// before the current group of 3 WMMAs so global loads overlap matrix work.
// ---------------------------------------------------------------------------
__global__ __launch_bounds__(128) void gru_gates_kernel(
    const bf16* __restrict__ Abf, int Kin, int lda,
    const int*  __restrict__ tok,
    const float* __restrict__ X2, int x_stride,   // Kin==2 f32 path when X2!=null
    const float* __restrict__ Wih2,               // [3H,2] f32 for X2 path
    const bf16* __restrict__ Wih, const float* __restrict__ bih,
    const bf16* __restrict__ Hp_bf, const float* __restrict__ Hp_f,
    const bf16* __restrict__ Whh, const float* __restrict__ bhh,
    float* __restrict__ Hn_f, bf16* __restrict__ Hn_bf,
    float* __restrict__ copy_f, int copy_stride)
{
    const int lane = threadIdx.x & 31;
    const int wave = threadIdx.x >> 5;
    const int m0   = wave * 16;            // 4 waves cover B=64 rows
    const int n0   = blockIdx.x * 16;      // column tile inside H
    const int half = lane >> 4;
    const int col  = lane & 15;
    const int row  = lane & 15;

    const v8f zf = {0.f, 0.f, 0.f, 0.f, 0.f, 0.f, 0.f, 0.f};
    v8f acc_i[3] = {zf, zf, zf};
    v8f acc_h[3] = {zf, zf, zf};

    // ---- input gates ----
    if (X2) {                               // K=2 elementwise (encoder layer 0)
#pragma unroll
        for (int g = 0; g < 3; ++g)
#pragma unroll
            for (int r = 0; r < 8; ++r) {
                const int m = m0 + r + half * 8;
                const int n = g * H + n0 + col;
                acc_i[g][r] = X2[m * x_stride + 0] * Wih2[n * 2 + 0]
                            + X2[m * x_stride + 1] * Wih2[n * 2 + 1];
            }
    } else {
        // Hoist gather: row base pointer is K-invariant.
        const bf16* arow = Abf
            + (size_t)(tok ? tok[m0 + row] : (m0 + row)) * lda;
        const bf16* brow[3];
#pragma unroll
        for (int g = 0; g < 3; ++g)
            brow[g] = Wih + (size_t)(g * H + n0 + col) * Kin + half * 16;

        Frag16 a_c, b_c[3];
        load_a_frag_row(a_c, arow, half);
#pragma unroll
        for (int g = 0; g < 3; ++g) load_b_frag_row(b_c[g], brow[g]);

        for (int k0 = 32; k0 < Kin; k0 += 32) {
            Frag16 a_n, b_n[3];
            load_a_frag_row(a_n, arow + k0, half);
#pragma unroll
            for (int g = 0; g < 3; ++g) {
                load_b_frag_row(b_n[g], brow[g] + k0);
                __builtin_prefetch(brow[g] + k0 + 32, 0, 0);
            }
#pragma unroll
            for (int g = 0; g < 3; ++g) acc_i[g] = wmma_bf16(a_c, b_c[g], acc_i[g]);
            a_c = a_n;
#pragma unroll
            for (int g = 0; g < 3; ++g) b_c[g] = b_n[g];
        }
#pragma unroll
        for (int g = 0; g < 3; ++g) acc_i[g] = wmma_bf16(a_c, b_c[g], acc_i[g]);
    }

    // ---- hidden gates (K = H), same 2-stage pipeline ----
    {
        const bf16* arow = Hp_bf + (size_t)(m0 + row) * H;
        const bf16* brow[3];
#pragma unroll
        for (int g = 0; g < 3; ++g)
            brow[g] = Whh + (size_t)(g * H + n0 + col) * H + half * 16;

        Frag16 a_c, b_c[3];
        load_a_frag_row(a_c, arow, half);
#pragma unroll
        for (int g = 0; g < 3; ++g) load_b_frag_row(b_c[g], brow[g]);

        for (int k0 = 32; k0 < H; k0 += 32) {
            Frag16 a_n, b_n[3];
            load_a_frag_row(a_n, arow + k0, half);
#pragma unroll
            for (int g = 0; g < 3; ++g) {
                load_b_frag_row(b_n[g], brow[g] + k0);
                __builtin_prefetch(brow[g] + k0 + 32, 0, 0);
            }
#pragma unroll
            for (int g = 0; g < 3; ++g) acc_h[g] = wmma_bf16(a_c, b_c[g], acc_h[g]);
            a_c = a_n;
#pragma unroll
            for (int g = 0; g < 3; ++g) b_c[g] = b_n[g];
        }
#pragma unroll
        for (int g = 0; g < 3; ++g) acc_h[g] = wmma_bf16(a_c, b_c[g], acc_h[g]);
    }

    // ---- GRU elementwise update on the D fragment ----
    const int n = n0 + col;
    const float bir = bih[n],         bhr = bhh[n];
    const float biz = bih[H + n],     bhz = bhh[H + n];
    const float bin = bih[2 * H + n], bhn = bhh[2 * H + n];
#pragma unroll
    for (int r = 0; r < 8; ++r) {
        const int m = m0 + r + half * 8;
        const float rg = sigm(acc_i[0][r] + bir + acc_h[0][r] + bhr);
        const float zg = sigm(acc_i[1][r] + biz + acc_h[1][r] + bhz);
        const float ng = tanhf(acc_i[2][r] + bin + rg * (acc_h[2][r] + bhn));
        const float hp = Hp_f[(size_t)m * H + n];
        const float hn = (1.f - zg) * ng + zg * hp;
        Hn_f[(size_t)m * H + n]  = hn;
        Hn_bf[(size_t)m * H + n] = (bf16)hn;
        if (copy_f) copy_f[(size_t)m * copy_stride + n] = hn;
    }
}

// ---------------------------------------------------------------------------
// Generic WMMA GEMM + bias: C[64,N] = A[64,K](bf16) @ W[N,K]^T + bias.
// Block 128 (4 waves = 4 M tiles), grid = N/16. 2-stage pipelined K loop.
// ---------------------------------------------------------------------------
__global__ __launch_bounds__(128) void gemm_bias_kernel(
    const bf16* __restrict__ A, int lda, int K,
    const bf16* __restrict__ W,
    const float* __restrict__ bias,
    float* __restrict__ Cf, int ldc,
    bf16* __restrict__ Cbf, int ldcb)
{
    const int lane = threadIdx.x & 31;
    const int wave = threadIdx.x >> 5;
    const int m0   = wave * 16;
    const int n0   = blockIdx.x * 16;
    const int half = lane >> 4;
    const int col  = lane & 15;

    const bf16* arow = A + (size_t)(m0 + (lane & 15)) * lda;
    const bf16* brow = W + (size_t)(n0 + col) * K + half * 16;

    v8f acc = {0.f, 0.f, 0.f, 0.f, 0.f, 0.f, 0.f, 0.f};
    Frag16 a_c, b_c;
    load_a_frag_row(a_c, arow, half);
    load_b_frag_row(b_c, brow);
    for (int k0 = 32; k0 < K; k0 += 32) {
        Frag16 a_n, b_n;
        load_a_frag_row(a_n, arow + k0, half);
        load_b_frag_row(b_n, brow + k0);
        __builtin_prefetch(brow + k0 + 32, 0, 0);
        acc = wmma_bf16(a_c, b_c, acc);
        a_c = a_n; b_c = b_n;
    }
    acc = wmma_bf16(a_c, b_c, acc);

    const int n = n0 + col;
    const float bn = bias ? bias[n] : 0.f;
#pragma unroll
    for (int r = 0; r < 8; ++r) {
        const int m = m0 + r + half * 8;
        const float v = acc[r] + bn;
        if (Cf)  Cf[(size_t)m * ldc + n]   = v;
        if (Cbf) Cbf[(size_t)m * ldcb + n] = (bf16)v;
    }
}

// scores[b,s] = dot(enc[b,s,:], q[b,:]) — one wave per (b,s)
__global__ __launch_bounds__(256) void attn_score_kernel(
    const float* __restrict__ enc, const float* __restrict__ q,
    float* __restrict__ sc)
{
    const int gw   = (blockIdx.x * blockDim.x + threadIdx.x) >> 5;
    const int lane = threadIdx.x & 31;
    const int b = gw >> 9;
    const int s = gw & (SS - 1);
    const float* e  = enc + ((size_t)b * SS + s) * H;
    const float* qq = q + (size_t)b * H;
    float acc = 0.f;
    for (int h = lane; h < H; h += 32) acc += e[h] * qq[h];
#pragma unroll
    for (int off = 16; off > 0; off >>= 1) acc += __shfl_xor(acc, off, 32);
    if (lane == 0) sc[(size_t)b * SS + s] = acc;
}

// in-place softmax over S per batch row; also scatters weights into attn[B,S,T]
__global__ __launch_bounds__(256) void softmax_kernel(
    float* __restrict__ sc, float* __restrict__ attn_out, int t)
{
    __shared__ float buf[256];
    const int b = blockIdx.x, tid = threadIdx.x;
    float* row = sc + (size_t)b * SS;

    float mx = -3.4e38f;
    for (int s = tid; s < SS; s += 256) mx = fmaxf(mx, row[s]);
    buf[tid] = mx; __syncthreads();
    for (int st = 128; st > 0; st >>= 1) {
        if (tid < st) buf[tid] = fmaxf(buf[tid], buf[tid + st]);
        __syncthreads();
    }
    mx = buf[0]; __syncthreads();

    float sum = 0.f;
    for (int s = tid; s < SS; s += 256) {
        const float e = __expf(row[s] - mx);
        row[s] = e; sum += e;
    }
    buf[tid] = sum; __syncthreads();
    for (int st = 128; st > 0; st >>= 1) {
        if (tid < st) buf[tid] += buf[tid + st];
        __syncthreads();
    }
    const float inv = 1.f / buf[0];
    for (int s = tid; s < SS; s += 256) {
        const float v = row[s] * inv;
        row[s] = v;
        attn_out[(size_t)b * SS * TT + (size_t)s * TT + t] = v;
    }
}

// concat[:, 0:512] = h1 (bf16 copy)
__global__ __launch_bounds__(256) void copy_h_concat_kernel(
    const bf16* __restrict__ h1bf, bf16* __restrict__ concat)
{
    const int idx = blockIdx.x * 256 + threadIdx.x;
    const int b = idx >> 9, h = idx & (H - 1);
    concat[(size_t)b * 1024 + h] = h1bf[(size_t)b * H + h];
}

// concat[:, 512:1024] = attention-weighted sum of enc outputs
__global__ __launch_bounds__(256) void attn_apply_kernel(
    const float* __restrict__ w, const float* __restrict__ enc,
    bf16* __restrict__ concat)
{
    const int idx = blockIdx.x * 256 + threadIdx.x;
    const int b = idx >> 9, h = idx & (H - 1);
    const float* wr = w + (size_t)b * SS;
    const float* e  = enc + (size_t)b * SS * H + h;
    float acc = 0.f;
    for (int s = 0; s < SS; ++s) acc += wr[s] * e[(size_t)s * H];
    concat[(size_t)b * 1024 + 512 + h] = (bf16)acc;
}

__global__ __launch_bounds__(256) void argmax_kernel(
    const float* __restrict__ logits, int ldl, int* __restrict__ tok)
{
    __shared__ float bv[256]; __shared__ int bi[256];
    const int b = blockIdx.x, tid = threadIdx.x;
    const float* row = logits + (size_t)b * ldl;
    float best = -3.4e38f; int bidx = 0;
    for (int d = tid; d < DD; d += 256) {
        const float v = row[d];
        if (v > best) { best = v; bidx = d; }
    }
    bv[tid] = best; bi[tid] = bidx; __syncthreads();
    for (int st = 128; st > 0; st >>= 1) {
        if (tid < st) {
            const float ov = bv[tid + st]; const int oi = bi[tid + st];
            if (ov > bv[tid] || (ov == bv[tid] && oi < bi[tid])) {
                bv[tid] = ov; bi[tid] = oi;
            }
        }
        __syncthreads();
    }
    if (tid == 0) tok[b] = bi[0];
}

__global__ void f2bf_kernel(const float* __restrict__ in, bf16* __restrict__ out, int n) {
    const int i = blockIdx.x * 256 + threadIdx.x;
    if (i < n) out[i] = (bf16)in[i];
}

__global__ void zero_init_kernel(float* h0f, bf16* h0b, float* h1f, bf16* h1b,
                                 int n, int* tok) {
    const int i = blockIdx.x * 256 + threadIdx.x;
    if (i < n) { h0f[i] = 0.f; h0b[i] = (bf16)0.f; h1f[i] = 0.f; h1b[i] = (bf16)0.f; }
    if (i < BB) tok[i] = 0;
}

__global__ void copy_hidden_kernel(const float* __restrict__ h0,
                                   const float* __restrict__ h1,
                                   float* __restrict__ outp) {
    const int i = blockIdx.x * 256 + threadIdx.x;
    if (i < BB * H) { outp[i] = h0[i]; outp[BB * H + i] = h1[i]; }
}

// ---------------------------------------------------------------------------
extern "C" void kernel_launch(void* const* d_in, const int* in_sizes, int n_in,
                              void* d_out, int out_size, void* d_ws, size_t ws_size,
                              hipStream_t stream)
{
    const float* x     = (const float*)d_in[0];
    const float* emb   = (const float*)d_in[1];
    const float* cWih0 = (const float*)d_in[2];
    const float* cWhh0 = (const float*)d_in[3];
    const float* cbih0 = (const float*)d_in[4];
    const float* cbhh0 = (const float*)d_in[5];
    const float* cWih1 = (const float*)d_in[6];
    const float* cWhh1 = (const float*)d_in[7];
    const float* cbih1 = (const float*)d_in[8];
    const float* cbhh1 = (const float*)d_in[9];
    const float* wWih0 = (const float*)d_in[10];
    const float* wWhh0 = (const float*)d_in[11];
    const float* wbih0 = (const float*)d_in[12];
    const float* wbhh0 = (const float*)d_in[13];
    const float* wWih1 = (const float*)d_in[14];
    const float* wWhh1 = (const float*)d_in[15];
    const float* wbih1 = (const float*)d_in[16];
    const float* wbhh1 = (const float*)d_in[17];
    const float* Wq    = (const float*)d_in[18];
    const float* bq    = (const float*)d_in[19];
    const float* Wc    = (const float*)d_in[20];
    const float* bc    = (const float*)d_in[21];
    const float* Wfc   = (const float*)d_in[22];
    const float* bfc   = (const float*)d_in[23];

    float* vec_out = (float*)d_out;                    // [B,T,D]
    float* hidden  = vec_out + (size_t)BB * TT * DD;   // [2,B,H]
    float* attn    = hidden + (size_t)2 * BB * H;      // [B,S,T]

    // ---- workspace carve (bump allocator, 256B aligned) ----
    char* wp = (char*)d_ws;
    auto alloc = [&](size_t bytes) -> char* {
        char* r = wp; wp += (bytes + 255) & ~(size_t)255; return r;
    };
    float* enc_f    = (float*)alloc((size_t)BB * SS * H * 4);
    bf16* emb_bf    = (bf16*)alloc((size_t)DD * H * 2);
    bf16* Wfc_bf    = (bf16*)alloc((size_t)DD * H * 2);
    bf16* cWhh0_bf  = (bf16*)alloc((size_t)3 * H * H * 2);
    bf16* cWih1_bf  = (bf16*)alloc((size_t)3 * H * H * 2);
    bf16* cWhh1_bf  = (bf16*)alloc((size_t)3 * H * H * 2);
    bf16* wWih0_bf  = (bf16*)alloc((size_t)3 * H * H * 2);
    bf16* wWhh0_bf  = (bf16*)alloc((size_t)3 * H * H * 2);
    bf16* wWih1_bf  = (bf16*)alloc((size_t)3 * H * H * 2);
    bf16* wWhh1_bf  = (bf16*)alloc((size_t)3 * H * H * 2);
    bf16* Wq_bf     = (bf16*)alloc((size_t)H * H * 2);
    bf16* Wc_bf     = (bf16*)alloc((size_t)H * 2 * H * 2);
    float* h0f[2]; bf16* h0b[2]; float* h1f[2]; bf16* h1b[2];
    for (int i = 0; i < 2; ++i) {
        h0f[i] = (float*)alloc((size_t)BB * H * 4);
        h0b[i] = (bf16*)alloc((size_t)BB * H * 2);
        h1f[i] = (float*)alloc((size_t)BB * H * 4);
        h1b[i] = (bf16*)alloc((size_t)BB * H * 2);
    }
    float* qf     = (float*)alloc((size_t)BB * H * 4);
    float* scw    = (float*)alloc((size_t)BB * SS * 4);
    bf16*  concat = (bf16*)alloc((size_t)BB * 1024 * 2);
    bf16*  combbf = (bf16*)alloc((size_t)BB * H * 2);
    int*   tok    = (int*)alloc((size_t)BB * 4);

    auto cvt = [&](const float* a, bf16* o, size_t n) {
        f2bf_kernel<<<(unsigned)((n + 255) / 256), 256, 0, stream>>>(a, o, (int)n);
    };
    cvt(emb,   emb_bf,   (size_t)DD * H);
    cvt(Wfc,   Wfc_bf,   (size_t)DD * H);
    cvt(cWhh0, cWhh0_bf, (size_t)3 * H * H);
    cvt(cWih1, cWih1_bf, (size_t)3 * H * H);
    cvt(cWhh1, cWhh1_bf, (size_t)3 * H * H);
    cvt(wWih0, wWih0_bf, (size_t)3 * H * H);
    cvt(wWhh0, wWhh0_bf, (size_t)3 * H * H);
    cvt(wWih1, wWih1_bf, (size_t)3 * H * H);
    cvt(wWhh1, wWhh1_bf, (size_t)3 * H * H);
    cvt(Wq,    Wq_bf,    (size_t)H * H);
    cvt(Wc,    Wc_bf,    (size_t)H * 2 * H);

    zero_init_kernel<<<(BB * H + 255) / 256, 256, 0, stream>>>(
        h0f[0], h0b[0], h1f[0], h1b[0], BB * H, tok);

    // ================= encoder: 512 sequential steps =================
    int p = 0;
    for (int t = 0; t < SS; ++t) {
        // layer 0: input dim 2 (elementwise gi), WMMA for gh
        gru_gates_kernel<<<H / 16, 128, 0, stream>>>(
            nullptr, 0, 0, nullptr,
            x + (size_t)t * 2, SS * 2, cWih0,
            nullptr, cbih0,
            h0b[p], h0f[p], cWhh0_bf, cbhh0,
            h0f[p ^ 1], h0b[p ^ 1], nullptr, 0);
        // layer 1: A = new h0; also scatter h1 into enc_outs[b,t,:]
        gru_gates_kernel<<<H / 16, 128, 0, stream>>>(
            h0b[p ^ 1], H, H, nullptr,
            nullptr, 0, nullptr,
            cWih1_bf, cbih1,
            h1b[p], h1f[p], cWhh1_bf, cbhh1,
            h1f[p ^ 1], h1b[p ^ 1], enc_f + (size_t)t * H, SS * H);
        p ^= 1;
    }
    // S=512 even -> p == 0 here; decoder continues from the same state.

    // ================= decoder: 32 greedy steps =================
    for (int t = 0; t < TT; ++t) {
        // layer 0: A = emb[tok] (gather folded into WMMA A load)
        gru_gates_kernel<<<H / 16, 128, 0, stream>>>(
            emb_bf, H, H, tok,
            nullptr, 0, nullptr,
            wWih0_bf, wbih0,
            h0b[p], h0f[p], wWhh0_bf, wbhh0,
            h0f[p ^ 1], h0b[p ^ 1], nullptr, 0);
        gru_gates_kernel<<<H / 16, 128, 0, stream>>>(
            h0b[p ^ 1], H, H, nullptr,
            nullptr, 0, nullptr,
            wWih1_bf, wbih1,
            h1b[p], h1f[p], wWhh1_bf, wbhh1,
            h1f[p ^ 1], h1b[p ^ 1], nullptr, 0);
        p ^= 1;

        // q = h1 @ Wq^T + bq
        gemm_bias_kernel<<<H / 16, 128, 0, stream>>>(
            h1b[p], H, H, Wq_bf, bq, qf, H, nullptr, 0);
        // scores / softmax (also writes attn[B,S,T])
        attn_score_kernel<<<(BB * SS) / 8, 256, 0, stream>>>(enc_f, qf, scw);
        softmax_kernel<<<BB, 256, 0, stream>>>(scw, attn, t);
        // concat = [h1, att]
        copy_h_concat_kernel<<<(BB * H) / 256, 256, 0, stream>>>(h1b[p], concat);
        attn_apply_kernel<<<(BB * H) / 256, 256, 0, stream>>>(scw, enc_f, concat);
        // comb = concat @ Wc^T + bc  (bf16 out feeds the logits GEMM)
        gemm_bias_kernel<<<H / 16, 128, 0, stream>>>(
            concat, 1024, 1024, Wc_bf, bc, nullptr, 0, combbf, H);
        // logits = comb @ Wfc^T + bfc -> straight into vec_out[b,t,:]
        gemm_bias_kernel<<<DD / 16, 128, 0, stream>>>(
            combbf, H, H, Wfc_bf, bfc,
            vec_out + (size_t)t * DD, TT * DD, nullptr, 0);
        argmax_kernel<<<BB, 256, 0, stream>>>(vec_out + (size_t)t * DD, TT * DD, tok);
    }

    copy_hidden_kernel<<<(BB * H + 255) / 256, 256, 0, stream>>>(h0f[p], h1f[p], hidden);
}
